// Graph_aspp_39359080301144
// MI455X (gfx1250) — compile-verified, matrix operand order checked by reading, not run on hardware
//
#include <hip/hip_runtime.h>
#include <math.h>

typedef __attribute__((ext_vector_type(16))) _Float16 v16h;
typedef __attribute__((ext_vector_type(8)))  _Float16 v8h;
typedef __attribute__((ext_vector_type(8)))  float    v8f;

#define LDSS 72   // LDS row stride in halves (144B = 16B-aligned, conflict-friendly)

__device__ __forceinline__ float sigmf(float x) { return 1.0f / (1.0f + __expf(-x)); }

// gfx1250 async global->LDS copy of 16 bytes per lane (ASYNCcnt-tracked).
__device__ __forceinline__ void async_ld_b128(unsigned lds_off, const void* gaddr)
{
    asm volatile("global_load_async_to_lds_b128 %0, %1, off"
                 :: "v"(lds_off), "v"(gaddr) : "memory");
}
__device__ __forceinline__ void wait_async0()
{
    asm volatile("s_wait_asynccnt 0x0" ::: "memory");
}

// ---------------------------------------------------------------------------
// Generic TN GEMM:  D[M,N] = sum_k A[m,k] * Bt[n,k]   (A: MxK row-major,
// Bt: NxK row-major), f16 operands, f32 accumulate via v_wmma_f32_16x16x32_f16.
// CONV=true: A is an implicit im2col of a [256pix x 1024ch] f16 image with
// 10 taps (9 dilated 3x3 taps + 1 center 1x1 tap), K = 10*1024; out-of-bounds
// taps stream from a zeroed scratch page so the whole stage stays async.
// Tile: 128(M) x 128(N) per workgroup; 8 waves as 4(m) x 2(n); each wave does
// 32x64 = 2x4 fragments; K-tile 64 = 2 sub-steps -> 16 v_wmma per stage.
// Double-buffered LDS: async DMA of tile t+1 overlaps WMMA on tile t.
// ---------------------------------------------------------------------------
template<bool CONV>
__global__ __launch_bounds__(256)
void k_gemm(const _Float16* __restrict__ A, long strideA, int lda,
            const _Float16* __restrict__ Bt, long strideB, int ldb,
            int M, int N, int K,
            const float* __restrict__ bias,          // per-N, or null
            int act,                                  // 0 none, 1 relu, 2 sigmoid
            const float* __restrict__ arow, long sArow, float ascale, // factor
            const float* __restrict__ res,  long sRes, int ldres,     // f32 residual
            float*    __restrict__ oF, long sOF, int ldoF,
            _Float16* __restrict__ oH, long sOH, int ldoH,
            const float* __restrict__ cscale, long sCs,               // per-N scale for o2
            _Float16* __restrict__ o2, long sO2, int ldo2,
            float* __restrict__ rsum, long sRs,       // fused row-sum (atomic), no stores
            const _Float16* __restrict__ zp,          // zeroed 16B source (conv OOB)
            int dil)
{
    __shared__ _Float16 sA[2][128 * LDSS];
    __shared__ _Float16 sB[2][128 * LDSS];

    const int b    = blockIdx.z;
    const int m0   = blockIdx.y * 128;
    const int n0   = blockIdx.x * 128;
    const int t    = threadIdx.x;
    const int lane = t & 31;
    const int w    = t >> 5;
    const int wm   = w & 3;        // 4 wave rows   (32 M each)
    const int wn   = w >> 2;       // 2 wave cols   (64 N each)
    const int lrow = lane & 15;
    const int lhalf = lane >> 4;

    const _Float16* Ab = A + (long)b * strideA;
    const _Float16* Bb = Bt + (long)b * strideB;

    // async-issue one 128x64-half A tile + B tile into buffer (kt & 1)
    auto issue = [&](int kt) {
        const int k0 = kt * 64;
        _Float16* dA = sA[kt & 1];
        _Float16* dB = sB[kt & 1];
        #pragma unroll
        for (int it = 0; it < 4; ++it) {
            int chunk = t + it * 256;         // 1024 chunks of 8 halves
            int row = chunk >> 3;
            int kc  = (chunk & 7) * 8;
            const _Float16* srcA;
            if (!CONV) {
                srcA = Ab + (long)(m0 + row) * lda + (k0 + kc);
            } else {
                int k = k0 + kc;
                int tap = k >> 10;            // 1024 channels per tap
                int c   = k & 1023;
                int p   = m0 + row;
                int y = p >> 4, x = p & 15;
                int sy = y, sx = x;
                bool ok = true;
                if (tap < 9) {
                    sy = y + (tap / 3 - 1) * dil;
                    sx = x + (tap % 3 - 1) * dil;
                    ok = ((unsigned)sy < 16u) && ((unsigned)sx < 16u);
                }
                srcA = ok ? (Ab + (long)(sy * 16 + sx) * lda + c) : zp;
            }
            async_ld_b128((unsigned)(size_t)(void*)&dA[row * LDSS + kc], srcA);
            __builtin_prefetch((const void*)(srcA + 64), 0, 0);
            const _Float16* srcB = Bb + (long)(n0 + row) * ldb + (k0 + kc);
            async_ld_b128((unsigned)(size_t)(void*)&dB[row * LDSS + kc], srcB);
            __builtin_prefetch((const void*)(srcB + 64), 0, 0);
        }
    };

    v8f acc[2][4] = {};
    const int T = K >> 6;                     // K-tiles of 64

    issue(0);
    for (int kt = 0; kt < T; ++kt) {
        wait_async0();                        // this wave's tile-kt copies done
        __syncthreads();                      // tile kt visible; buf (kt+1)&1 free
        if (kt + 1 < T) issue(kt + 1);        // overlap DMA with compute

        const _Float16* a  = sA[kt & 1];
        const _Float16* bs = sB[kt & 1];
        #pragma unroll
        for (int ks = 0; ks < 64; ks += 32) {
            const int klo = ks + lhalf * 8;
            const int khi = ks + 16 + lhalf * 8;
            v16h af[2], bf[4];
            #pragma unroll
            for (int mf = 0; mf < 2; ++mf) {
                int r = wm * 32 + mf * 16 + lrow;
                v8h lo = *(const v8h*)&a[r * LDSS + klo];
                v8h hi = *(const v8h*)&a[r * LDSS + khi];
                af[mf] = __builtin_shufflevector(lo, hi, 0,1,2,3,4,5,6,7,8,9,10,11,12,13,14,15);
            }
            #pragma unroll
            for (int nf = 0; nf < 4; ++nf) {
                int r = wn * 64 + nf * 16 + lrow;
                v8h lo = *(const v8h*)&bs[r * LDSS + klo];
                v8h hi = *(const v8h*)&bs[r * LDSS + khi];
                bf[nf] = __builtin_shufflevector(lo, hi, 0,1,2,3,4,5,6,7,8,9,10,11,12,13,14,15);
            }
            #pragma unroll
            for (int mf = 0; mf < 2; ++mf)
                #pragma unroll
                for (int nf = 0; nf < 4; ++nf)
                    acc[mf][nf] = __builtin_amdgcn_wmma_f32_16x16x32_f16(
                        false, af[mf], false, bf[nf], (short)0, acc[mf][nf], false, false);
        }
    }

    // ---- epilogue
    const float*    arow_b = arow ? arow + (long)b * sArow : nullptr;
    const float*    res_b  = res  ? res  + (long)b * sRes  : nullptr;
    float*          oF_b   = oF   ? oF   + (long)b * sOF   : nullptr;
    _Float16*       oH_b   = oH   ? oH   + (long)b * sOH   : nullptr;
    const float*    cs_b   = cscale ? cscale + (long)b * sCs : nullptr;
    _Float16*       o2_b   = o2   ? o2   + (long)b * sO2   : nullptr;
    float*          rs_b   = rsum ? rsum + (long)b * sRs   : nullptr;

    #pragma unroll
    for (int nf = 0; nf < 4; ++nf) {
        int n = n0 + wn * 64 + nf * 16 + lrow;
        float bn = bias ? bias[n] : 0.0f;
        float cs = cs_b ? cs_b[n] : 0.0f;
        #pragma unroll
        for (int mf = 0; mf < 2; ++mf) {
            #pragma unroll
            for (int i = 0; i < 8; ++i) {
                int m = m0 + wm * 32 + mf * 16 + lhalf * 8 + i;
                float v = acc[mf][nf][i] + bn;
                if (act == 1)      v = fmaxf(v, 0.0f);
                else if (act == 2) v = sigmf(v);
                if (rs_b) {
                    float s = v;  // row-sum over the 16 columns of this fragment
                    s += __shfl_xor(s, 1);
                    s += __shfl_xor(s, 2);
                    s += __shfl_xor(s, 4);
                    s += __shfl_xor(s, 8);
                    if (lrow == 0) atomicAdd(&rs_b[m], s);
                } else {
                    float fac = ascale * (arow_b ? arow_b[m] : 1.0f);
                    float ov  = (res_b ? res_b[(long)m * ldres + n] : 0.0f) + fac * v;
                    if (oF_b) oF_b[(long)m * ldoF + n] = ov;
                    if (oH_b) oH_b[(long)m * ldoH + n] = (_Float16)ov;
                    if (o2_b) o2_b[(long)m * ldo2 + n] = (_Float16)(ov * cs);
                }
            }
        }
    }
}

// ---------------------------------------------------------------------------
// Small helper kernels
// ---------------------------------------------------------------------------

// x_sum = x_cur * upsample2x_align_corners(x_lat) + x_cur   (NCHW in,
// pixel-major [b][p][c] f32+f16 out)
__global__ void k_xsum(const float* __restrict__ xcur, const float* __restrict__ xlat,
                       float* __restrict__ oF, _Float16* __restrict__ oH)
{
    int bp = blockIdx.x;            // b*256 + p
    int b = bp >> 8, p = bp & 255;
    int y = p >> 4, x = p & 15;
    float fy = (float)y * (7.0f / 15.0f);
    float fx = (float)x * (7.0f / 15.0f);
    int y0 = (int)floorf(fy), x0 = (int)floorf(fx);
    int y1 = min(y0 + 1, 7), x1 = min(x0 + 1, 7);
    float wy = fy - (float)y0, wx = fx - (float)x0;
    for (int c = threadIdx.x; c < 1024; c += 256) {
        const float* lat = xlat + ((long)(b * 1024 + c)) * 64;
        float v00 = lat[y0 * 8 + x0], v01 = lat[y0 * 8 + x1];
        float v10 = lat[y1 * 8 + x0], v11 = lat[y1 * 8 + x1];
        float top = v00 * (1.0f - wx) + v01 * wx;
        float bot = v10 * (1.0f - wx) + v11 * wx;
        float up  = top * (1.0f - wy) + bot * wy;
        float xc  = xcur[((long)(b * 1024 + c)) * 256 + p];
        float s   = xc * up + xc;
        long o = (long)bp * 1024 + c;
        oF[o] = s;
        oH[o] = (_Float16)s;
    }
}

// per-(b,c) mean & max over 256 pixels of f [b][p][c]
__global__ void k_reduce_avgmax(const float* __restrict__ f,
                                float* __restrict__ avg, float* __restrict__ mx)
{
    int b = blockIdx.x;
    int c = threadIdx.x;            // 256 threads
    const float* fb = f + (long)b * 256 * 256;
    float s = 0.0f, m = -3.4e38f;
    for (int p = 0; p < 256; ++p) {
        float v = fb[(long)p * 256 + c];
        s += v;
        m = fmaxf(m, v);
    }
    avg[b * 256 + c] = s * (1.0f / 256.0f);
    mx[b * 256 + c]  = m;
}

// cw = sigmoid(relu(avg@avg_w.T) + relu(mx@max_w.T))   (per batch, 128 threads)
__global__ void k_cw(const float* __restrict__ avg, const float* __restrict__ mx,
                     const float* __restrict__ aw, const float* __restrict__ mw,
                     float* __restrict__ cw)
{
    int b = blockIdx.x;
    int m = threadIdx.x;            // 128 threads
    const float* a = avg + b * 256;
    const float* x = mx + b * 256;
    const float* awr = aw + m * 256;
    const float* mwr = mw + m * 256;
    float ra = 0.0f, rm = 0.0f;
    for (int c = 0; c < 256; ++c) {
        ra += a[c] * awr[c];
        rm += x[c] * mwr[c];
    }
    cw[b * 128 + m] = sigmf(fmaxf(ra, 0.0f) + fmaxf(rm, 0.0f));
}

__global__ void k_zero(float* __restrict__ p, long n)
{
    long i = (long)blockIdx.x * blockDim.x + threadIdx.x;
    if (i < n) p[i] = 0.0f;
}

__global__ void k_rsqrt(float* __restrict__ dvec, const float* __restrict__ dsum, long n)
{
    long i = (long)blockIdx.x * blockDim.x + threadIdx.x;
    if (i < n) dvec[i] = rsqrtf(dsum[i]);
}

// out[b][n][m] = in[b][m][n] * (scale ? scale[b][m] : 1)   (f16 -> f16)
__global__ void k_transpose_f16(const _Float16* __restrict__ in, long strideIn,
                                int M, int N,
                                const float* __restrict__ scale, long sScale,
                                _Float16* __restrict__ out, long strideOut)
{
    __shared__ _Float16 tile[32][33];
    __shared__ float srow[32];
    int b = blockIdx.z;
    int m0 = blockIdx.y * 32, n0 = blockIdx.x * 32;
    int tx = threadIdx.x, ty = threadIdx.y;      // (32, 8)
    const _Float16* inb = in + (long)b * strideIn;
    for (int i = ty; i < 32; i += 8)
        tile[i][tx] = inb[(long)(m0 + i) * N + n0 + tx];
    if (ty == 0) srow[tx] = scale ? scale[(long)b * sScale + m0 + tx] : 1.0f;
    __syncthreads();
    _Float16* outb = out + (long)b * strideOut;
    for (int i = ty; i < 32; i += 8)
        outb[(long)(n0 + i) * M + m0 + tx] =
            (_Float16)((float)tile[tx][i] * srow[tx]);
}

// pixel-major [b][p][o=1024] f32  ->  NCHW [b][o][p]
__global__ void k_pm_to_nchw(const float* __restrict__ pm, float* __restrict__ out)
{
    int bo = blockIdx.x;            // b*1024 + o
    int b = bo >> 10, o = bo & 1023;
    int p = threadIdx.x;            // 256 threads
    out[(long)bo * 256 + p] = pm[((long)b * 256 + p) * 1024 + o];
}

// weight prep: wcat[r][t][o][c]  (t<9: rate{r}_w[o][c][ky][kx], t==9: conv_w[o][c])
__global__ void k_prep_wcat(const float* __restrict__ r0, const float* __restrict__ r1,
                            const float* __restrict__ r2, const float* __restrict__ cw,
                            _Float16* __restrict__ wcat, long n)
{
    long i = (long)blockIdx.x * blockDim.x + threadIdx.x;
    if (i >= n) return;
    int c = (int)(i & 1023);
    int o = (int)((i >> 10) & 255);
    int t = (int)((i >> 18) % 10);
    int r = (int)(i / (10L << 18));
    float v;
    if (t < 9) {
        const float* rw = (r == 0) ? r0 : (r == 1) ? r1 : r2;
        v = rw[((long)(o * 1024 + c)) * 9 + t];
    } else {
        v = cw[(long)o * 1024 + c];
    }
    wcat[i] = (_Float16)v;
}

__global__ void k_prep_bias3(const float* __restrict__ rb0, const float* __restrict__ rb1,
                             const float* __restrict__ rb2, const float* __restrict__ cb,
                             float* __restrict__ b3)
{
    int i = blockIdx.x * blockDim.x + threadIdx.x;
    if (i >= 3 * 256) return;
    int r = i >> 8, o = i & 255;
    const float* rb = (r == 0) ? rb0 : (r == 1) ? rb1 : rb2;
    b3[i] = rb[o] + cb[o];
}

__global__ void k_f32_to_f16(_Float16* __restrict__ dst, const float* __restrict__ src, long n)
{
    long i = (long)blockIdx.x * blockDim.x + threadIdx.x;
    if (i < n) dst[i] = (_Float16)src[i];
}

// gcnT[j][c] = gcn_w[c][j]   (256x256)
__global__ void k_prep_gcnT(const float* __restrict__ g, _Float16* __restrict__ o)
{
    int i = blockIdx.x * blockDim.x + threadIdx.x;
    if (i >= 256 * 256) return;
    int j = i >> 8, c = i & 255;
    o[i] = (_Float16)g[c * 256 + j];
}

// ---------------------------------------------------------------------------
// Host orchestration
// ---------------------------------------------------------------------------
static void gemm(hipStream_t s, bool conv,
                 const _Float16* A, long strideA, int lda,
                 const _Float16* Bt, long strideB, int ldb,
                 int M, int N, int K, int batch,
                 const float* bias, int act,
                 const float* arow, long sArow, float ascale,
                 const float* res, long sRes, int ldres,
                 float* oF, long sOF, int ldoF,
                 _Float16* oH, long sOH, int ldoH,
                 const float* cscale, long sCs, _Float16* o2, long sO2, int ldo2,
                 float* rsum, long sRs, const _Float16* zp, int dil)
{
    dim3 g(N / 128, M / 128, batch), blk(256);
    if (conv)
        k_gemm<true><<<g, blk, 0, s>>>(A, strideA, lda, Bt, strideB, ldb, M, N, K,
            bias, act, arow, sArow, ascale, res, sRes, ldres,
            oF, sOF, ldoF, oH, sOH, ldoH, cscale, sCs, o2, sO2, ldo2, rsum, sRs, zp, dil);
    else
        k_gemm<false><<<g, blk, 0, s>>>(A, strideA, lda, Bt, strideB, ldb, M, N, K,
            bias, act, arow, sArow, ascale, res, sRes, ldres,
            oF, sOF, ldoF, oH, sOH, ldoH, cscale, sCs, o2, sO2, ldo2, rsum, sRs, zp, dil);
}

extern "C" void kernel_launch(void* const* d_in, const int* in_sizes, int n_in,
                              void* d_out, int out_size, void* d_ws, size_t ws_size,
                              hipStream_t stream)
{
    const int B = 64, C = 1024, CQ = 256, MI = 128, HW = 256;

    const float* x_cur   = (const float*)d_in[0];
    const float* x_lat   = (const float*)d_in[1];
    const float* conv_w  = (const float*)d_in[2];
    const float* conv_b  = (const float*)d_in[3];
    const float* conv1_w = (const float*)d_in[4];
    const float* conv1_b = (const float*)d_in[5];
    const float* conv3_w = (const float*)d_in[6];
    const float* conv3_b = (const float*)d_in[7];
    const float* rate_w[3] = {(const float*)d_in[8], (const float*)d_in[10], (const float*)d_in[12]};
    const float* rate_b[3] = {(const float*)d_in[9], (const float*)d_in[11], (const float*)d_in[13]};
    const float* ck_w  = (const float*)d_in[14];
    const float* ck_b  = (const float*)d_in[15];
    const float* avg_w = (const float*)d_in[16];
    const float* max_w = (const float*)d_in[17];
    const float* gcn_w = (const float*)d_in[18];
    (void)in_sizes; (void)n_in; (void)out_size; (void)ws_size;

    // ---- workspace bump allocator (256B aligned)
    char* wp = (char*)d_ws;
    auto alloc = [&](size_t bytes) -> void* {
        void* p = (void*)wp;
        wp += (bytes + 255) & ~(size_t)255;
        return p;
    };
    float*     xsumF  = (float*)alloc((size_t)B * HW * C * 4);
    _Float16*  xsumH  = (_Float16*)alloc((size_t)B * HW * C * 2);
    float*     featF  = (float*)alloc((size_t)B * HW * C * 4);
    _Float16*  featH  = (_Float16*)alloc((size_t)B * HW * C * 2);
    float*     fF     = (float*)alloc((size_t)B * HW * CQ * 4);
    _Float16*  fH     = (_Float16*)alloc((size_t)B * HW * CQ * 2);
    _Float16*  qH     = (_Float16*)alloc((size_t)B * HW * MI * 2);
    _Float16*  qsH    = (_Float16*)alloc((size_t)B * HW * MI * 2);
    _Float16*  qTH    = (_Float16*)alloc((size_t)B * MI * HW * 2);
    _Float16*  XdTH   = (_Float16*)alloc((size_t)B * CQ * HW * 2);
    _Float16*  PtXTH  = (_Float16*)alloc((size_t)B * CQ * MI * 2);
    _Float16*  LXH    = (_Float16*)alloc((size_t)B * HW * CQ * 2);
    _Float16*  catH   = (_Float16*)alloc((size_t)B * HW * 3 * CQ * 2);
    float*     avg    = (float*)alloc((size_t)B * CQ * 4);
    float*     mx     = (float*)alloc((size_t)B * CQ * 4);
    float*     cw     = (float*)alloc((size_t)B * MI * 4);
    float*     dsum   = (float*)alloc((size_t)B * HW * 4);
    float*     dvec   = (float*)alloc((size_t)B * HW * 4);
    _Float16*  wcat   = (_Float16*)alloc((size_t)3 * 10 * CQ * C * 2);
    float*     bias3  = (float*)alloc((size_t)3 * CQ * 4);
    _Float16*  c1w16  = (_Float16*)alloc((size_t)C * CQ * 2);
    _Float16*  c3w16  = (_Float16*)alloc((size_t)C * 3 * CQ * 2);
    _Float16*  ckw16  = (_Float16*)alloc((size_t)MI * CQ * 2);
    _Float16*  gcnT16 = (_Float16*)alloc((size_t)CQ * CQ * 2);
    _Float16*  zpage  = (_Float16*)alloc(256);   // zeroed 16B source for OOB taps
    float*     outpm  = xsumF;  // alias: x_sum is dead before the final GEMM

    // ---- weight prep + zero page
    {
        long n = 3L * 10 * CQ * C;
        k_prep_wcat<<<(unsigned)((n + 255) / 256), 256, 0, stream>>>(
            rate_w[0], rate_w[1], rate_w[2], conv_w, wcat, n);
        k_prep_bias3<<<3, 256, 0, stream>>>(rate_b[0], rate_b[1], rate_b[2], conv_b, bias3);
        k_f32_to_f16<<<(C * CQ + 255) / 256, 256, 0, stream>>>(c1w16, conv1_w, (long)C * CQ);
        k_f32_to_f16<<<(C * 3 * CQ + 255) / 256, 256, 0, stream>>>(c3w16, conv3_w, (long)C * 3 * CQ);
        k_f32_to_f16<<<(MI * CQ + 255) / 256, 256, 0, stream>>>(ckw16, ck_w, (long)MI * CQ);
        k_prep_gcnT<<<(CQ * CQ + 255) / 256, 256, 0, stream>>>(gcn_w, gcnT16);
        k_zero<<<1, 64, 0, stream>>>((float*)zpage, 64);
    }

    // ---- x_sum
    k_xsum<<<B * HW, 256, 0, stream>>>(x_cur, x_lat, xsumF, xsumH);

    const long sPM  = (long)HW * C;     // per-batch stride, pixel-major C=1024
    const long sPQ  = (long)HW * CQ;    // pixel-major Cq=256
    const long sPM2 = (long)HW * MI;    // pixel-major Mi=128

    const int dils[3] = {1, 2, 4};
    for (int g = 0; g < 3; ++g) {
        const _Float16* src16 = (g == 0) ? xsumH : featH;
        if (g > 0) {
            // feature = conv1x1(f, conv1_w, conv1_b) + x_sum
            gemm(stream, false, fH, sPQ, CQ, c1w16, 0, CQ, HW, C, CQ, B,
                 conv1_b, 0, nullptr, 0, 1.0f, xsumF, sPM, C,
                 featF, sPM, C, featH, sPM, C,
                 nullptr, 0, nullptr, 0, 0, nullptr, 0, zpage, 0);
        }
        // f = conv3x3_dilated(src) + conv1x1(src, conv_w) + (rate_b + conv_b)
        gemm(stream, true, src16, sPM, C, wcat + (long)g * 10 * CQ * C, 0, C,
             HW, CQ, 10 * C, B,
             bias3 + g * CQ, 0, nullptr, 0, 1.0f, nullptr, 0, 0,
             fF, sPQ, CQ, fH, sPQ, CQ,
             nullptr, 0, nullptr, 0, 0, nullptr, 0, zpage, dils[g]);

        // ---- graph reasoning
        k_reduce_avgmax<<<B, 256, 0, stream>>>(fF, avg, mx);
        k_cw<<<B, 128, 0, stream>>>(avg, mx, avg_w, max_w, cw);

        // q = relu(ftr @ ck_w^T + ck_b); qs = q * cw[n]  (fused second output)
        gemm(stream, false, fH, sPQ, CQ, ckw16, 0, CQ, HW, MI, CQ, B,
             ck_b, 1, nullptr, 0, 1.0f, nullptr, 0, 0,
             nullptr, 0, 0, qH, sPM2, MI,
             cw, MI, qsH, sPM2, MI, nullptr, 0, zpage, 0);

        // d-sums: rowsum_j sigmoid(sum_m q[i,m]*qs[j,m])  (A never materialized)
        k_zero<<<(B * HW + 255) / 256, 256, 0, stream>>>(dsum, (long)B * HW);
        gemm(stream, false, qH, sPM2, MI, qsH, sPM2, MI, HW, HW, MI, B,
             nullptr, 2, nullptr, 0, 1.0f, nullptr, 0, 0,
             nullptr, 0, 0, nullptr, 0, 0,
             nullptr, 0, nullptr, 0, 0, dsum, HW, zpage, 0);
        k_rsqrt<<<(B * HW + 255) / 256, 256, 0, stream>>>(dvec, dsum, (long)B * HW);

        // XdT[c][j] = f[j][c] * d[j] ; qT[m][j] = q[j][m]
        {
            dim3 blkT(32, 8);
            dim3 g1(CQ / 32, HW / 32, B);
            k_transpose_f16<<<g1, blkT, 0, stream>>>(fH, sPQ, HW, CQ, dvec, HW, XdTH, (long)CQ * HW);
            dim3 g2(MI / 32, HW / 32, B);
            k_transpose_f16<<<g2, blkT, 0, stream>>>(qH, sPM2, HW, MI, nullptr, 0, qTH, (long)MI * HW);
        }

        // PtXT[c][m] = sum_j XdT[c][j] * qT[m][j]
        gemm(stream, false, XdTH, (long)CQ * HW, HW, qTH, (long)MI * HW, HW,
             CQ, MI, HW, B,
             nullptr, 0, nullptr, 0, 1.0f, nullptr, 0, 0,
             nullptr, 0, 0, PtXTH, (long)CQ * MI, MI,
             nullptr, 0, nullptr, 0, 0, nullptr, 0, zpage, 0);

        // LX[j][c] = f[j][c] - d[j] * sum_m qs[j,m] * PtXT[c][m]
        gemm(stream, false, qsH, sPM2, MI, PtXTH, (long)CQ * MI, MI,
             HW, CQ, MI, B,
             nullptr, 0, dvec, HW, -1.0f, fF, sPQ, CQ,
             nullptr, 0, 0, LXH, sPQ, CQ,
             nullptr, 0, nullptr, 0, 0, nullptr, 0, zpage, 0);

        // Y[j][c2] = f[j][c2] + sum_c LX[j,c] * gcnT[c2][c]   -> cat slice g
        gemm(stream, false, LXH, sPQ, CQ, gcnT16, 0, CQ, HW, CQ, CQ, B,
             nullptr, 0, nullptr, 0, 1.0f, fF, sPQ, CQ,
             nullptr, 0, 0, catH + (long)g * CQ, (long)HW * 3 * CQ, 3 * CQ,
             nullptr, 0, nullptr, 0, 0, nullptr, 0, zpage, 0);
    }

    // ---- final 1x1 conv over cat: out[p][o] = cat[p][:] . conv3_w[o][:] + conv3_b
    gemm(stream, false, catH, (long)HW * 3 * CQ, 3 * CQ, c3w16, 0, 3 * CQ,
         HW, C, 3 * CQ, B,
         conv3_b, 0, nullptr, 0, 1.0f, nullptr, 0, 0,
         outpm, sPM, C, nullptr, 0, 0,
         nullptr, 0, nullptr, 0, 0, nullptr, 0, zpage, 0);

    // ---- pixel-major -> NCHW
    k_pm_to_nchw<<<B * C, 256, 0, stream>>>(outpm, (float*)d_out);
}
